// FNO1d_11785390260350
// MI455X (gfx1250) — matrix operand order repaired and checked by reading, not compiled, not run on hardware
//
#include <hip/hip_runtime.h>
#include <math.h>

// ---------------------------------------------------------------------------
// FNO1d forward for MI455X (gfx1250, wave32, WMMA bf16 16x16x32 f32-accum,
// TDM tensor_load_to_lds + global_load_async_to_lds staging, double-buffered)
// B=32, N=8192, C=64, MODES=64, NL=4, PROJ_H=128
// ---------------------------------------------------------------------------

typedef unsigned short u16;
typedef unsigned long long u64;
typedef __attribute__((ext_vector_type(16))) __bf16 v16bf;
typedef __attribute__((ext_vector_type(8)))  float  v8f;
typedef __attribute__((ext_vector_type(4)))  unsigned int v4u;
typedef __attribute__((ext_vector_type(8)))  int v8i;
typedef __attribute__((ext_vector_type(4)))  int v4i;

union FragU { v16bf v; unsigned int u[8]; };

__device__ __forceinline__ u16 f2bf(float f) {
    union { float f; unsigned int u; } v; v.f = f;
    unsigned int u = v.u;
    u += 0x7FFFu + ((u >> 16) & 1u);        // round-to-nearest-even
    return (u16)(u >> 16);
}

__device__ __forceinline__ float geluf(float x) {
    return 0.5f * x * (1.0f + erff(x * 0.70710678118654752f));
}

// --- CDNA5 async global->LDS copies (ASYNCcnt), GVS mode: SGPR base + v32 off
__device__ __forceinline__ unsigned lds_off(const void* p) {
    return (unsigned)(u64)p;                // shared aperture: addr[31:0] = LDS offset
}
__device__ __forceinline__ void async_b128(void* lds_dst, const void* gbase, unsigned byte_off) {
    asm volatile("global_load_async_to_lds_b128 %0, %1, %2"
                 :: "v"(lds_off(lds_dst)), "v"(byte_off), "s"((u64)gbase) : "memory");
}
__device__ __forceinline__ void async_b64(void* lds_dst, const void* gbase, unsigned byte_off) {
    asm volatile("global_load_async_to_lds_b64 %0, %1, %2"
                 :: "v"(lds_off(lds_dst)), "v"(byte_off), "s"((u64)gbase) : "memory");
}
__device__ __forceinline__ void wait_async0() {
    asm volatile("s_wait_asynccnt 0x0" ::: "memory");
}
__device__ __forceinline__ void wait_async6() {   // up to 6 async ops may remain outstanding
    asm volatile("s_wait_asynccnt 0x6" ::: "memory");
}

// --- CDNA5 TDM: 2D bf16 tensor tile -> LDS, row padding 128B->144B (pitch 72 bf16)
// tensor: dim0 = 8192 (n, contiguous), dim1 = 128 (c), stride0 = 8192
// tile:   dim0 = 64  (n chunk),        dim1 = 128 (all c)
__device__ __forceinline__ void tdm_load_basis_tile(const u16* gsrc, void* lds_dst) {
    u64 ga = (u64)gsrc;
    v4u g0;
    g0.x = 1u;                                        // count=1 (valid user descr)
    g0.y = lds_off(lds_dst);                          // lds_addr (bytes)
    g0.z = (unsigned)(ga & 0xFFFFFFFFu);              // global_addr[31:0]
    g0.w = (unsigned)((ga >> 32) & 0x1FFFFFFu) | (2u << 30);   // addr[56:32] | type=2
    v8i g1;
    g1[0] = (1 << 16)      // data_size = 2 bytes
          | (1 << 20)      // pad_enable
          | (4 << 22)      // pad_interval: 32 dwords (128B)
          | (3 << 25);     // pad_amount: 4 dwords (16B)
    g1[1] = (8192 & 0xFFFF) << 16;   // tensor_dim0[15:0] @ bits 63:48
    g1[2] = (128 & 0xFFFF) << 16;    // tensor_dim0[31:16]=0 | tensor_dim1[15:0] @ 95:80
    g1[3] = (64 & 0xFFFF) << 16;     // tensor_dim1[31:16]=0 | tile_dim0 @ 127:112
    g1[4] = 128;                     // tile_dim1 @ 143:128, tile_dim2=0
    g1[5] = 8192;                    // tensor_dim0_stride[31:0] @ 191:160
    g1[6] = 0;                       // stride0[47:32]=0 | tensor_dim1_stride[15:0]
    g1[7] = 0;
    v4i gz = {};
#if defined(__clang_major__) && (__clang_major__ >= 23)
    v8i gz8 = {};
    __builtin_amdgcn_tensor_load_to_lds(g0, g1, gz, gz, gz8, 0);
#else
    __builtin_amdgcn_tensor_load_to_lds(g0, g1, gz, gz, 0);
#endif
}

// A-operand fragment (16x32 bf16, stored [M][K] in LDS; pitchd = dword pitch)
__device__ __forceinline__ v16bf load_a_frag(const unsigned int* lds, int m0, int ks,
                                             int pitchd, int lane) {
    FragU f;
    int m  = m0 + (lane & 15);
    int o4 = (lane >> 4) * 4;
    const unsigned int* row = lds + m * pitchd + (ks >> 1);
#pragma unroll
    for (int j = 0; j < 4; ++j) f.u[j]     = row[o4 + j];
#pragma unroll
    for (int j = 0; j < 4; ++j) f.u[4 + j] = row[8 + o4 + j];
    return f.v;
}

// B-operand fragment (32x16 bf16, stored [N][K] in LDS)
__device__ __forceinline__ v16bf load_b_frag(const unsigned int* lds, int n0, int ks,
                                             int pitchd, int lane) {
    FragU f;
    int n = n0 + (lane & 15);
    int o = (lane >> 4) * 4;
    const unsigned int* row = lds + n * pitchd + (ks >> 1) + o;
#pragma unroll
    for (int j = 0; j < 4; ++j) f.u[j] = row[j];
#pragma unroll
    for (int j = 0; j < 4; ++j) f.u[4 + j] = row[8 + j];
    return f.v;
}

// ---------------------------------------------------------------------------
// 1) Trig basis tables (bf16): Tt[c][n] (fwd), T[n][c] (inverse)
//    c=2k -> cos(2*pi*k*n/N), c=2k+1 -> sin(2*pi*k*n/N)
// ---------------------------------------------------------------------------
__global__ void fno_basis_kernel(u16* __restrict__ Tt, u16* __restrict__ T) {
    int g = blockIdx.x * blockDim.x + threadIdx.x;
    int n = g & 8191;
    int c = g >> 13;
    if (c >= 128) return;
    int k = c >> 1;
    int m = (k * n) & 8191;
    float ang = (float)m * (6.283185307179586f / 8192.0f);
    float s, co;
    __sincosf(ang, &s, &co);
    u16 bv = f2bf((c & 1) ? s : co);
    Tt[(size_t)c * 8192 + n] = bv;
    T[(size_t)n * 128 + c]   = bv;
}

// ---------------------------------------------------------------------------
// 2) Lift (affine in n): h[b][c][n] = base(b,c) + slope(c)*n
// ---------------------------------------------------------------------------
__global__ void fno_lift_kernel(const float* __restrict__ x, const float* __restrict__ lw,
                                const float* __restrict__ lb, u16* __restrict__ h) {
    int r = blockIdx.x;
    int b = r >> 6, c = r & 63;
    float base = lb[c];
#pragma unroll
    for (int d = 0; d < 5; ++d) base += x[b * 5 + d] * lw[c * 6 + d];
    float slope = lw[c * 6 + 5] * (1.0f / 8191.0f);
    u16* row = h + (size_t)r * 8192;
    for (int n = threadIdx.x; n < 8192; n += blockDim.x)
        row[n] = f2bf(base + slope * (float)n);
}

// ---------------------------------------------------------------------------
// 3) Forward truncated DFT as GEMM: X[r][c] = sum_n h[r][n] * Tt[c][n]
//    Double-buffered: TDM (basis tile) + async b128 (h rows) for chunk k+1
//    issued before computing chunk k; partial-counter waits
//    (s_wait_asynccnt 6 / s_wait_tensorcnt 1) expose the overlap.
// ---------------------------------------------------------------------------
__global__ __launch_bounds__(256) void fno_fwd_dft_kernel(const u16* __restrict__ h,
                                                          const u16* __restrict__ Tt,
                                                          float* __restrict__ X) {
    constexpr int PA = 36, PB = 36;          // dword pitches (72 bf16)
    __shared__ unsigned int lA[2][64 * PA];
    __shared__ unsigned int lB[2][128 * PB]; // TDM pad: 128B row + 16B -> 144B = 36 dw
    int tid = threadIdx.x, lane = tid & 31, w = tid >> 5;
    int row0 = blockIdx.x * 64;
    int rb = w & 3, cq = w >> 2;
    int m = tid >> 2, q = tid & 3;           // A staging coords
    unsigned gbase = (unsigned)(((row0 + m) * 8192 + q * 16) * 2);
    v8f acc[4] = {};

    // prologue: issue chunk 0 into buffer 0
    if (w == 0) tdm_load_basis_tile(Tt, lB[0]);
    {
        unsigned int* dst = lA[0] + m * PA + q * 8;
        async_b128(dst, h, gbase);
        async_b128(dst + 4, h, gbase + 16);
    }

    int cur = 0;
    for (int k0 = 0; k0 < 8192; k0 += 64) {
        int nxt = cur ^ 1;
        if (k0 + 64 < 8192) {
            // issue chunk k0+64 into the other buffer (consumed two barriers ago)
            if (w == 0) tdm_load_basis_tile(Tt + k0 + 64, lB[nxt]);
            unsigned goff = gbase + (unsigned)((k0 + 64) * 2);
            unsigned int* dst = lA[nxt] + m * PA + q * 8;
            async_b128(dst, h, goff);
            async_b128(dst + 4, h, goff + 16);
            wait_async6();                               // chunks are 2 ops: up to 3 chunks may still be in flight
            if (w == 0) __builtin_amdgcn_s_wait_tensorcnt(1);
        } else {
            wait_async0();
            if (w == 0) __builtin_amdgcn_s_wait_tensorcnt(0);
        }
        __syncthreads();
#pragma unroll
        for (int ks = 0; ks < 64; ks += 32) {
            v16bf a = load_a_frag(lA[cur], rb * 16, ks, PA, lane);
            v16bf bfr[4];
#pragma unroll
            for (int t = 0; t < 4; ++t)
                bfr[t] = load_b_frag(lB[cur], (cq * 4 + t) * 16, ks, PB, lane);
#pragma unroll
            for (int t = 0; t < 4; ++t)
                acc[t] = __builtin_amdgcn_wmma_f32_16x16x32_bf16(
                    false, a, false, bfr[t], (short)0, acc[t], false, false);
        }
        __syncthreads();                                 // buffer free for refill
        cur = nxt;
    }
    int mrow = (lane >> 4) << 3, ncol = lane & 15;
#pragma unroll
    for (int t = 0; t < 4; ++t) {
        int cb = cq * 4 + t;
#pragma unroll
        for (int v = 0; v < 8; ++v)
            X[(size_t)(row0 + rb * 16 + v + mrow) * 128 + cb * 16 + ncol] = acc[t][v];
    }
}

// ---------------------------------------------------------------------------
// 4) Per-mode complex channel mixing + irfft scale folding (async-staged).
// ---------------------------------------------------------------------------
__global__ __launch_bounds__(256) void fno_mode_mix_kernel(const float* __restrict__ X,
                                                           const float* __restrict__ spW,
                                                           u16* __restrict__ coef) {
    __shared__ float lx[2048 * 2];
    __shared__ float lw[64 * 64 * 2];
    int k = blockIdx.x, tid = threadIdx.x;
    for (int r = tid; r < 2048; r += 256)
        async_b64(&lx[2 * r], X, (unsigned)((r * 128 + 2 * k) * 4));
    for (int e = tid; e < 4096; e += 256) {
        int i = e >> 6, o = e & 63;
        async_b64(&lw[2 * e], spW, (unsigned)((((i * 64 + o) * 64 + k) * 2) * 4));
    }
    wait_async0();
    __syncthreads();
    float ck = (k == 0) ? (1.0f / 8192.0f) : (2.0f / 8192.0f);
    for (int u = 0; u < 8; ++u) {
        int e = tid * 8 + u;
        int b = e >> 6, o = e & 63;
        float re = 0.f, im = 0.f;
        for (int i = 0; i < 64; ++i) {
            float xc = lx[2 * (b * 64 + i)], xs = lx[2 * (b * 64 + i) + 1];
            float wr = lw[2 * (i * 64 + o)], wi = lw[2 * (i * 64 + o) + 1];
            re += xc * wr + xs * wi;       // Re(xf*w), xf = Xc - i*Xs
            im += xc * wi - xs * wr;       // Im(xf*w)
        }
        coef[(size_t)e * 128 + 2 * k]     = f2bf(ck * re);
        coef[(size_t)e * 128 + 2 * k + 1] = f2bf(-ck * im);
    }
}

// ---------------------------------------------------------------------------
// 5) Fused inverse DFT + 1x1 conv + bias + exact GELU as ONE K=192 GEMM.
// ---------------------------------------------------------------------------
__global__ __launch_bounds__(256) void fno_inv_fuse_kernel(const u16* __restrict__ coef,
                                                           const u16* __restrict__ T,
                                                           const u16* __restrict__ h_in,
                                                           const float* __restrict__ wcw,
                                                           const float* __restrict__ wcb,
                                                           u16* __restrict__ h_out) {
    constexpr int PAd = 100, PB = 100;       // dword pitch (200 bf16)
    __shared__ unsigned int lA[64 * PAd];    // [o][K=192]
    __shared__ unsigned int lB[128 * PB];    // [n][K=192]
    int tid = threadIdx.x, lane = tid & 31, w = tid >> 5;
    int b = blockIdx.y;
    int nbase = blockIdx.x * 128;

    {   // A part1: spectral coefficients (bf16) -> 4x b128 async
        int o = tid >> 2, p = tid & 3;
        unsigned goff = (unsigned)(((b * 64 + o) * 128 + p * 32) * 2);
        unsigned int* dst = lA + o * PAd + p * 16;
#pragma unroll
        for (int j = 0; j < 4; ++j) async_b128(dst + 4 * j, coef, goff + 16 * j);
    }
    {   // B part1: basis rows T[n][0:128] -> 8x b128 async
        int tn = tid >> 1, hf = tid & 1;
        unsigned goff = (unsigned)(((nbase + tn) * 128 + hf * 64) * 2);
        unsigned int* dst = lB + tn * PB + hf * 32;
#pragma unroll
        for (int j = 0; j < 8; ++j) async_b128(dst + 4 * j, T, goff + 16 * j);
    }
    {   // A part2: wc weights f32 -> bf16 at K offset 128 (VALU, overlaps async)
        int o = tid >> 2, i0 = (tid & 3) * 16;
        const float4* src = (const float4*)(wcw + o * 64 + i0);
        unsigned int* dst = lA + o * PAd + 64 + (i0 >> 1);
#pragma unroll
        for (int j = 0; j < 4; ++j) {
            float4 v = src[j];
            dst[2 * j]     = (unsigned int)f2bf(v.x) | ((unsigned int)f2bf(v.y) << 16);
            dst[2 * j + 1] = (unsigned int)f2bf(v.z) | ((unsigned int)f2bf(v.w) << 16);
        }
    }
    {   // B part2: h transpose [i][n] -> [n][128+i] (b16 DS stores)
        int tn = tid & 127, ih = tid >> 7;
        u16* lBh = (u16*)lB;
        for (int i = ih * 32; i < ih * 32 + 32; ++i)
            lBh[tn * (2 * PB) + 128 + i] = h_in[(size_t)(b * 64 + i) * 8192 + nbase + tn];
    }
    wait_async0();
    __syncthreads();

    int rb = w & 3, cq = w >> 2;
    v8f acc[4] = {};
#pragma unroll
    for (int ks = 0; ks < 192; ks += 32) {
        v16bf a = load_a_frag(lA, rb * 16, ks, PAd, lane);
        v16bf bfr[4];
#pragma unroll
        for (int j = 0; j < 4; ++j)
            bfr[j] = load_b_frag(lB, (cq * 4 + j) * 16, ks, PB, lane);
#pragma unroll
        for (int j = 0; j < 4; ++j)
            acc[j] = __builtin_amdgcn_wmma_f32_16x16x32_bf16(
                false, a, false, bfr[j], (short)0, acc[j], false, false);
    }
    int mrow = (lane >> 4) << 3, ncol = lane & 15;
#pragma unroll
    for (int j = 0; j < 4; ++j) {
        int n = nbase + (cq * 4 + j) * 16 + ncol;
#pragma unroll
        for (int v = 0; v < 8; ++v) {
            int o = rb * 16 + v + mrow;
            float val = acc[j][v] + wcb[o];
            h_out[(size_t)(b * 64 + o) * 8192 + n] = f2bf(geluf(val));
        }
    }
}

// ---------------------------------------------------------------------------
// 6) Projection GEMM + fused GELU + 128->1 reduction (LDS atomics)
// ---------------------------------------------------------------------------
__global__ __launch_bounds__(256) void fno_proj_kernel(const u16* __restrict__ h,
                                                       const float* __restrict__ p1w,
                                                       const float* __restrict__ p1b,
                                                       const float* __restrict__ p2w,
                                                       const float* __restrict__ p2b,
                                                       float* __restrict__ out) {
    constexpr int PA = 36, PB = 36;
    __shared__ unsigned int lA[128 * PA];
    __shared__ unsigned int lB[128 * PB];
    __shared__ float ysum[128];
    int tid = threadIdx.x, lane = tid & 31, w = tid >> 5;
    int b = blockIdx.y;
    int nbase = blockIdx.x * 128;
    if (tid < 128) ysum[tid] = 0.f;

    {   // stage A: p1_w f32 [128][64] -> bf16
        int j = tid >> 1, i0 = (tid & 1) * 32;
        const float4* src = (const float4*)(p1w + j * 64 + i0);
        unsigned int* dst = lA + j * PA + (i0 >> 1);
#pragma unroll
        for (int qq = 0; qq < 8; ++qq) {
            float4 v = src[qq];
            dst[2 * qq]     = (unsigned int)f2bf(v.x) | ((unsigned int)f2bf(v.y) << 16);
            dst[2 * qq + 1] = (unsigned int)f2bf(v.z) | ((unsigned int)f2bf(v.w) << 16);
        }
    }
    {   // stage B: h transpose [i][n] -> [n][i]
        int tn = tid & 127, ih = tid >> 7;
        u16* lBh = (u16*)lB;
        for (int i = ih * 32; i < ih * 32 + 32; ++i)
            lBh[tn * (2 * PB) + i] = h[(size_t)(b * 64 + i) * 8192 + nbase + tn];
    }
    __syncthreads();

    v8f acc[8] = {};
#pragma unroll
    for (int ks = 0; ks < 64; ks += 32) {
        v16bf a = load_a_frag(lA, w * 16, ks, PA, lane);
        v16bf bfr[8];
#pragma unroll
        for (int cb = 0; cb < 8; ++cb)
            bfr[cb] = load_b_frag(lB, cb * 16, ks, PB, lane);
#pragma unroll
        for (int cb = 0; cb < 8; ++cb)
            acc[cb] = __builtin_amdgcn_wmma_f32_16x16x32_bf16(
                false, a, false, bfr[cb], (short)0, acc[cb], false, false);
    }
    int mrow = (lane >> 4) << 3, ncol = lane & 15;
#pragma unroll
    for (int cb = 0; cb < 8; ++cb) {
        float part = 0.f;
#pragma unroll
        for (int v = 0; v < 8; ++v) {
            int j = w * 16 + v + mrow;
            float g = geluf(acc[cb][v] + p1b[j]);
            part += g * p2w[j];
        }
        atomicAdd(&ysum[cb * 16 + ncol], part);
    }
    __syncthreads();
    if (tid < 128) out[(size_t)b * 8192 + nbase + tid] = ysum[tid] + p2b[0];
}

// ---------------------------------------------------------------------------
// Host orchestration (all on `stream`; workspace only, no allocation)
// ---------------------------------------------------------------------------
extern "C" void kernel_launch(void* const* d_in, const int* in_sizes, int n_in,
                              void* d_out, int out_size, void* d_ws, size_t ws_size,
                              hipStream_t stream) {
    (void)in_sizes; (void)n_in; (void)out_size; (void)ws_size;
    const float* x      = (const float*)d_in[0];
    const float* lift_w = (const float*)d_in[1];
    const float* lift_b = (const float*)d_in[2];
    const float* sp_W   = (const float*)d_in[3];
    const float* wc_w   = (const float*)d_in[4];
    const float* wc_b   = (const float*)d_in[5];
    const float* p1_w   = (const float*)d_in[6];
    const float* p1_b   = (const float*)d_in[7];
    const float* p2_w   = (const float*)d_in[8];
    const float* p2_b   = (const float*)d_in[9];

    char* ws = (char*)d_ws;
    u16*   h0   = (u16*)(ws);                                   // 32 MB
    u16*   h1   = (u16*)(ws + (size_t)32 * 1024 * 1024);        // 32 MB
    u16*   Tt   = (u16*)(ws + (size_t)64 * 1024 * 1024);        //  2 MB [c][n]
    u16*   T    = (u16*)(ws + (size_t)66 * 1024 * 1024);        //  2 MB [n][c]
    float* X    = (float*)(ws + (size_t)68 * 1024 * 1024);      //  1 MB
    u16*   coef = (u16*)(ws + (size_t)69 * 1024 * 1024);        // 0.5 MB

    fno_basis_kernel<<<4096, 256, 0, stream>>>(Tt, T);
    fno_lift_kernel<<<2048, 256, 0, stream>>>(x, lift_w, lift_b, h0);

    u16* hin = h0; u16* hout = h1;
    for (int l = 0; l < 4; ++l) {
        fno_fwd_dft_kernel<<<32, 256, 0, stream>>>(hin, Tt, X);
        fno_mode_mix_kernel<<<64, 256, 0, stream>>>(X, sp_W + (size_t)l * 64 * 64 * 64 * 2, coef);
        fno_inv_fuse_kernel<<<dim3(64, 32), 256, 0, stream>>>(
            coef, T, hin, wc_w + (size_t)l * 4096, wc_b + (size_t)l * 64, hout);
        u16* tmp = hin; hin = hout; hout = tmp;
    }
    fno_proj_kernel<<<dim3(64, 32), 256, 0, stream>>>(hin, p1_w, p1_b, p2_w, p2_b, (float*)d_out);
}
